// AttentionPooler_15015205667482
// MI455X (gfx1250) — matrix operand order repaired and checked by reading, not compiled, not run on hardware
//
#include <hip/hip_runtime.h>
#include <hip/hip_bf16.h>

// ---- problem constants (from reference) ----
#define LAYER   7
#define NB      16
#define LSEQ    512
#define DDIM    768          // K of big GEMM
#define HDIM    256          // N of big GEMM
#define MROWS   (NB*LSEQ)    // 8192 = M of big GEMM
#define MAXSPAN 32

// ---- tiling ----
#define BM 128               // rows per block (16 per wave, 8 waves)
#define BN 64                // cols per block (4 x 16-wide WMMA tiles per wave)
#define BK 32                // one 16x16x32 bf16 WMMA K-step
#define NKCH (DDIM/BK)       // 24 K-chunks
#define NTILES (HDIM/16)     // 16 N-tiles total
#define ASTRIDE 36           // f32 per LDS A row (32 + 4 pad -> conflict-free, 16B aligned)

// packed-B workspace geometry (uints; one uint = 2 bf16 along K)
#define BP_UINTS (NTILES*NKCH*32*8)    // 98304 uints = 384 KB per (hi|lo) half
#define BP_HALF_BYTES (BP_UINTS*4)

typedef __attribute__((ext_vector_type(16))) __bf16        v16bf;
typedef __attribute__((ext_vector_type(8)))  float         v8f;
typedef __attribute__((ext_vector_type(8)))  unsigned int  v8u;

__device__ __forceinline__ unsigned bpack(__bf16 a, __bf16 b) {
    return ((unsigned)__builtin_bit_cast(unsigned short, b) << 16) |
            (unsigned)__builtin_bit_cast(unsigned short, a);
}

// async DMA: global (SGPR base + VGPR byte offset) -> LDS (VGPR byte offset), 16B/lane
__device__ __forceinline__ void async_lds_b128(unsigned ldsOff, unsigned gOff,
                                               unsigned long long gBase) {
    asm volatile("global_load_async_to_lds_b128 %0, %1, %2"
                 :: "v"(ldsOff), "v"(gOff), "s"(gBase) : "memory");
}

// ---------------------------------------------------------------------------
// Kernel 0: pack W_in (768x256 f32) into WMMA-fragment-ready bf16 hi/lo pairs.
// Layout: uint[(ntile*NKCH + kc)*256 + lane*8 + v] ; K = kc*32 + (lane>>4)*16 + 2v,
// N = ntile*16 + (lane&15).  hi half first, lo half at +BP_UINTS.
// ---------------------------------------------------------------------------
__global__ __launch_bounds__(32)
void pack_B_kernel(const float* __restrict__ W_in, unsigned* __restrict__ Bp)
{
    const int ntile = blockIdx.x;          // 0..15
    const int kc    = blockIdx.y;          // 0..23
    const int lane  = threadIdx.x;         // 0..31
    const int halfL = lane >> 4;
    const int l16   = lane & 15;
    const unsigned baseIdx = ((unsigned)(ntile * NKCH + kc)) * 256u + lane * 8u;
    #pragma unroll
    for (int v = 0; v < 8; ++v) {
        const int K = kc * BK + halfL * 16 + 2 * v;
        const int N = ntile * 16 + l16;
        const float x0 = W_in[(size_t)K * HDIM + N];
        const float x1 = W_in[(size_t)(K + 1) * HDIM + N];
        const __bf16 h0 = (__bf16)x0, h1 = (__bf16)x1;
        const __bf16 g0 = (__bf16)(x0 - (float)h0);
        const __bf16 g1 = (__bf16)(x1 - (float)h1);
        Bp[baseIdx + v]            = bpack(h0, h1);
        Bp[BP_UINTS + baseIdx + v] = bpack(g0, g1);
    }
}

// ---------------------------------------------------------------------------
// Kernel 1: proj[8192,256] = hs[8192,768] @ W_in[768,256] + b_in
// bf16 hi/lo split (acc += Ah*Bh + Al*Bh + Ah*Bl), async double-buffered LDS.
// ---------------------------------------------------------------------------
__global__ __launch_bounds__(256)
void proj_gemm_kernel(const float* __restrict__ hs_full,
                      const unsigned* __restrict__ Bp,
                      const float* __restrict__ b_in,
                      float* __restrict__ proj)
{
    const float* __restrict__ hs = hs_full + (size_t)LAYER * MROWS * DDIM;

    __shared__ float    As[2][BM][ASTRIDE];   // f32 A tiles (double buffered)
    __shared__ unsigned Bs[2][2][4 * 32 * 8]; // [buf][hi/lo][tile*256+lane*8+v]

    const int tid   = threadIdx.x;
    const int wave  = tid >> 5;
    const int lane  = tid & 31;
    const int halfL = lane >> 4;
    const int l16   = lane & 15;

    const int mBase     = blockIdx.x * BM;
    const int nBase     = blockIdx.y * BN;
    const int nTileBase = nBase >> 4;

    const unsigned long long hsBase = (unsigned long long)(uintptr_t)hs;
    const unsigned long long bpBase = (unsigned long long)(uintptr_t)Bp;
    const unsigned aLds = (unsigned)(uintptr_t)&As[0][0][0];
    const unsigned bLds = (unsigned)(uintptr_t)&Bs[0][0][0];

    // ---- async staging of one K-chunk into buffer b ----
    auto stage = [&](int kc2, int b) {
        // A: 128 rows x 32 f32 (128B/row) -> LDS rows of 144B
        #pragma unroll
        for (int i = 0; i < 4; ++i) {
            const int id  = tid + i * 256;          // 0..1023
            const int row = id >> 3;
            const int seg = id & 7;
            const unsigned gOff = ((unsigned)((mBase + row) * DDIM + kc2 * BK + seg * 4)) * 4u;
            const unsigned lOff = aLds + (unsigned)(b * BM * ASTRIDE + row * ASTRIDE) * 4u
                                       + (unsigned)seg * 16u;
            async_lds_b128(lOff, gOff, hsBase);
        }
        // B: 2 halves (hi/lo) x 4 tiles x 256 uints, linear both sides
        #pragma unroll
        for (int i = 0; i < 2; ++i) {
            const int id   = tid + i * 256;         // 0..511
            const int hilo = id >> 8;
            const int rem  = id & 255;              // b128 segment within half
            const int t    = rem >> 6;
            const int sub  = rem & 63;
            const unsigned gOff = (((unsigned)(nTileBase + t) * NKCH + kc2) * 256u + sub * 4u) * 4u
                                  + (unsigned)hilo * (unsigned)BP_HALF_BYTES;
            const unsigned lOff = bLds + ((unsigned)(b * 2 + hilo) * 1024u + (unsigned)rem * 4u) * 4u;
            async_lds_b128(lOff, gOff, bpBase);
        }
    };

    v8f acc[4] = {};

    stage(0, 0);                                    // prologue
    for (int kc = 0; kc < NKCH; ++kc) {
        const int buf = kc & 1;
        if (kc + 1 < NKCH) {
            stage(kc + 1, buf ^ 1);
            asm volatile("s_wait_asynccnt 0x6" ::: "memory");  // cur done, 6 next in flight
        } else {
            asm volatile("s_wait_asynccnt 0x0" ::: "memory");
        }
        __syncthreads();

        // ---- A fragments: read f32 runs, convert to bf16 hi/lo in registers ----
        const int arow = wave * 16 + l16;
        const float4* ap0 = (const float4*)&As[buf][arow][halfL * 16];
        const float4* ap1 = (const float4*)&As[buf][arow][halfL * 16 + 8];
        const float4 f0 = ap0[0], f1 = ap0[1], f2 = ap1[0], f3 = ap1[1];
        const float r[16] = {f0.x,f0.y,f0.z,f0.w, f1.x,f1.y,f1.z,f1.w,
                             f2.x,f2.y,f2.z,f2.w, f3.x,f3.y,f3.z,f3.w};
        v8u ahu, alu;
        #pragma unroll
        for (int v = 0; v < 8; ++v) {
            const float x0 = r[2 * v], x1 = r[2 * v + 1];
            const __bf16 h0 = (__bf16)x0, h1 = (__bf16)x1;
            const __bf16 g0 = (__bf16)(x0 - (float)h0);
            const __bf16 g1 = (__bf16)(x1 - (float)h1);
            ahu[v] = bpack(h0, h1);
            alu[v] = bpack(g0, g1);
        }
        const v16bf afh = __builtin_bit_cast(v16bf, ahu);
        const v16bf afl = __builtin_bit_cast(v16bf, alu);

        // ---- 4 N-tiles, 3 WMMAs each ----
        #pragma unroll
        for (int t = 0; t < 4; ++t) {
            const uint4* bph = (const uint4*)&Bs[buf][0][(t * 32 + lane) * 8];
            const uint4* bpl = (const uint4*)&Bs[buf][1][(t * 32 + lane) * 8];
            const uint4 bh0 = bph[0], bh1 = bph[1];
            const uint4 bl0 = bpl[0], bl1 = bpl[1];
            v8u bhu, blu;
            bhu[0]=bh0.x; bhu[1]=bh0.y; bhu[2]=bh0.z; bhu[3]=bh0.w;
            bhu[4]=bh1.x; bhu[5]=bh1.y; bhu[6]=bh1.z; bhu[7]=bh1.w;
            blu[0]=bl0.x; blu[1]=bl0.y; blu[2]=bl0.z; blu[3]=bl0.w;
            blu[4]=bl1.x; blu[5]=bl1.y; blu[6]=bl1.z; blu[7]=bl1.w;
            const v16bf bfh = __builtin_bit_cast(v16bf, bhu);
            const v16bf bfl = __builtin_bit_cast(v16bf, blu);
            acc[t] = __builtin_amdgcn_wmma_f32_16x16x32_bf16(false, afh, false, bfh, (short)0, acc[t], false, false);
            acc[t] = __builtin_amdgcn_wmma_f32_16x16x32_bf16(false, afl, false, bfh, (short)0, acc[t], false, false);
            acc[t] = __builtin_amdgcn_wmma_f32_16x16x32_bf16(false, afh, false, bfl, (short)0, acc[t], false, false);
        }
        __syncthreads();
    }

    // ---- epilogue: C layout (VGPR r: lanes0-15 -> M=r, lanes16-31 -> M=r+8) ----
    #pragma unroll
    for (int t = 0; t < 4; ++t) {
        const int col = nBase + t * 16 + l16;
        const float bias = b_in[col];
        #pragma unroll
        for (int r = 0; r < 8; ++r) {
            const int m = mBase + wave * 16 + (halfL ? r + 8 : r);
            proj[(size_t)m * HDIM + col] = acc[t][r] + bias;
        }
    }
}

// ---------------------------------------------------------------------------
// Kernel 2: per-span masked softmax over scores = proj @ w_score, then
// pooled[s,h] = sum_l att[l] * proj[row_l, h].  One 256-thread block per span.
// ---------------------------------------------------------------------------
__global__ __launch_bounds__(256)
void span_pool_kernel(const float* __restrict__ proj,
                      const int*   __restrict__ spans,
                      const float* __restrict__ w_score,
                      float* __restrict__ out)
{
    __shared__ float sSc[MAXSPAN];
    __shared__ float sAtt[MAXSPAN];

    const int s    = blockIdx.x;
    const int tid  = threadIdx.x;
    const int wave = tid >> 5;
    const int lane = tid & 31;

    const int bidx = spans[3 * s + 0];
    const int a    = spans[3 * s + 1];
    const int bEnd = spans[3 * s + 2];
    const int len  = bEnd - a;                 // 1..32
    const int rowBase = bidx * LSEQ + a;

    for (int l = wave; l < len; l += 8) {
        const float* pr = proj + (size_t)(rowBase + l) * HDIM;
        float partial = 0.f;
        #pragma unroll
        for (int u = 0; u < 8; ++u) {
            const int h = lane + 32 * u;
            partial += pr[h] * w_score[h];
        }
        #pragma unroll
        for (int off = 16; off > 0; off >>= 1)
            partial += __shfl_xor(partial, off, 32);
        if (lane == 0) sSc[l] = partial;
    }
    __syncthreads();

    if (tid < 32) {
        const float v = (tid < len) ? sSc[tid] : -3.0e38f;
        float m = v;
        #pragma unroll
        for (int off = 16; off > 0; off >>= 1)
            m = fmaxf(m, __shfl_xor(m, off, 32));
        const float e = (tid < len) ? __expf(v - m) : 0.f;
        float sum = e;
        #pragma unroll
        for (int off = 16; off > 0; off >>= 1)
            sum += __shfl_xor(sum, off, 32);
        sAtt[tid] = e / sum;
    }
    __syncthreads();

    float accv = 0.f;
    for (int l = 0; l < len; ++l)
        accv += sAtt[l] * proj[(size_t)(rowBase + l) * HDIM + tid];
    out[(size_t)s * HDIM + tid] = accv;
}

// ---------------------------------------------------------------------------
extern "C" void kernel_launch(void* const* d_in, const int* in_sizes, int n_in,
                              void* d_out, int out_size, void* d_ws, size_t ws_size,
                              hipStream_t stream)
{
    const float* hs      = (const float*)d_in[0];  // (13,16,512,768) f32
    const int*   spans   = (const int*)  d_in[1];  // (S,3) i32
    const float* W_in    = (const float*)d_in[2];  // (768,256) f32
    const float* b_in    = (const float*)d_in[3];  // (256,) f32
    const float* w_score = (const float*)d_in[4];  // (256,) f32
    float*       out     = (float*)d_out;          // (S,256) f32

    float*    proj = (float*)d_ws;                        // 8192*256 f32 = 8 MB
    unsigned* Bp   = (unsigned*)d_ws + (size_t)MROWS*HDIM; // packed B hi|lo (768 KB)

    const int nS = in_sizes[1] / 3;                // 1024 spans

    pack_B_kernel<<<dim3(NTILES, NKCH), 32, 0, stream>>>(W_in, Bp);
    proj_gemm_kernel<<<dim3(MROWS / BM, HDIM / BN), 256, 0, stream>>>(hs, Bp, b_in, proj);
    span_pool_kernel<<<nS, 256, 0, stream>>>(proj, spans, w_score, out);
}
